// TransformerBlock_953482739761
// MI455X (gfx1250) — compile-verified, matrix operand order checked by reading, not run on hardware
//
#include <hip/hip_runtime.h>

// ---------------------------------------------------------------------------
// MI455X (gfx1250) transformer block.
// bf16 WMMA (v_wmma_f32_16x16x32_bf16) for all GEMMs + flash attention,
// f32 accumulation. Working set (<200MB) fits the 192MB L2 -> compute bound.
// GEMM tiles are DMA'd into LDS with the Tensor Data Mover
// (tensor_load_to_lds + s_wait_tensorcnt), double-buffered so DMA overlaps
// WMMA; weights are pre-transposed once so both A and B tiles are identical
// row-major 128x32 bf16 tiles, and TDM padding (4dw per 16dw) reproduces the
// bank-conflict-free 40-element LDS row stride.
// ---------------------------------------------------------------------------

typedef __attribute__((ext_vector_type(16))) __bf16 v16bf;
typedef __attribute__((ext_vector_type(8)))  __bf16 v8bf;
typedef __attribute__((ext_vector_type(8)))  float  v8f;

#define BB   2
#define SS   2048
#define DD   1024
#define FF   4096
#define HH   16
#define DH   64
#define ROWS (BB * SS)   // 4096

#if defined(__AMDGCN__) && __has_builtin(__builtin_amdgcn_tensor_load_to_lds) && \
    __has_builtin(__builtin_amdgcn_s_wait_tensorcnt)
#define USE_TDM 1
#else
#define USE_TDM 0
#endif

__device__ __forceinline__ v16bf load_frag(const __bf16* p0, const __bf16* p1) {
  union { v16bf v; v8bf h[2]; } u;
  u.h[0] = *(const v8bf*)p0;
  u.h[1] = *(const v8bf*)p1;
  return u.v;
}

#if USE_TDM
typedef __attribute__((ext_vector_type(4))) unsigned int tdm_u32x4;
typedef __attribute__((ext_vector_type(8))) int          tdm_i32x8;
typedef __attribute__((ext_vector_type(4))) int          tdm_i32x4;

// 2D bf16 tile DMA: global (tileRows x tileK, row stride rowStrideElems) ->
// LDS at ldsByte, packed with +4dw pad per 16dw => 40-elem LDS row stride.
// D# layout per CDNA5 ISA 8.3/8.4 (group0: count/lds/global/type; group1:
// data_size, pad, tensor dims = tile dims (tiles fully in-bounds), strides).
// This toolchain exposes the 6-arg builtin (clang-23 form):
//   (uint32x4 g0, int32x8 g1, int32x4 g2, int32x4 g3, int32x8 extra, i32 cpol)
__device__ __forceinline__ void tdm_load_tile_bf16(const __bf16* gsrc, unsigned ldsByte,
                                                   unsigned tileK, unsigned tileRows,
                                                   unsigned rowStrideElems) {
  unsigned long long ga = (unsigned long long)(__UINTPTR_TYPE__)gsrc;
  unsigned long long s0 = (unsigned long long)rowStrideElems;
  tdm_u32x4 g0;
  g0[0] = 1u;                                               // count=1, user mode
  g0[1] = ldsByte;                                          // lds_addr
  g0[2] = (unsigned)(ga & 0xFFFFFFFFu);                     // global_addr lo
  g0[3] = (unsigned)((ga >> 32) & 0x1FFFFFFu) | (2u << 30); // global_addr hi | type=2
  tdm_i32x8 g1;
  g1[0] = (int)((1u << 16) | (1u << 20) | (3u << 22) | (3u << 25)); // 2B, pad 4dw/16dw
  g1[1] = (int)((tileK    & 0xFFFFu) << 16);   // tensor_dim0 lo16 (hi bits 0)
  g1[2] = (int)((tileRows & 0xFFFFu) << 16);   // tensor_dim0 hi=0 | tensor_dim1 lo16
  g1[3] = (int)((tileK    & 0xFFFFu) << 16);   // tensor_dim1 hi=0 | tile_dim0
  g1[4] = (int)( tileRows & 0xFFFFu);          // tile_dim1 | tile_dim2=0
  g1[5] = (int)(s0 & 0xFFFFFFFFu);             // tensor_dim0_stride lo32
  g1[6] = (int)((s0 >> 32) & 0xFFFFu);         // stride0 hi16 | stride1 lo16 = 0
  g1[7] = 0;                                   // stride1 hi
  tdm_i32x4 gz4 = {0, 0, 0, 0};
  tdm_i32x8 gz8 = {0, 0, 0, 0, 0, 0, 0, 0};
  __builtin_amdgcn_tensor_load_to_lds(g0, g1, gz4, gz4, gz8, 0);
}
#endif

// ---------------------------------------------------------------------------
// f32 -> bf16 conversion (4 elems / thread)
// ---------------------------------------------------------------------------
__global__ __launch_bounds__(256) void cvt_f32_bf16(const float* __restrict__ in,
                                                    __bf16* __restrict__ out, int n) {
  int base = (blockIdx.x * 256 + threadIdx.x) * 4;
  if (base + 3 < n) {
    float4 f = *(const float4*)(in + base);
    out[base + 0] = (__bf16)f.x;
    out[base + 1] = (__bf16)f.y;
    out[base + 2] = (__bf16)f.z;
    out[base + 3] = (__bf16)f.w;
  } else {
    for (int i = base; i < n; ++i) out[i] = (__bf16)in[i];
  }
}

// ---------------------------------------------------------------------------
// Transpose + convert: in f32 [Kdim x Ndim] -> out bf16 [Ndim x Kdim]
// ---------------------------------------------------------------------------
__global__ __launch_bounds__(256) void transpose_cvt_bf16(const float* __restrict__ in,
                                                          __bf16* __restrict__ out,
                                                          int Kdim, int Ndim) {
  __shared__ float tile[32][33];
  const int kb = blockIdx.x * 32, nb = blockIdx.y * 32;
  const int tx = threadIdx.x & 31, ty = threadIdx.x >> 5;  // 8 row-groups
  for (int r = ty; r < 32; r += 8)
    tile[r][tx] = in[(size_t)(kb + r) * Ndim + nb + tx];
  __syncthreads();
  for (int r = ty; r < 32; r += 8)
    out[(size_t)(nb + r) * Kdim + kb + tx] = (__bf16)tile[tx][r];
}

// ---------------------------------------------------------------------------
// Tiled bf16 WMMA GEMM: C[M,N] = A[M,K] @ B[K,N] (+bias)(ReLU), B given as
// Bt[N,K] (pre-transposed). 256 threads = 8 waves; tile 128x128, K-step 32.
// Wave grid 4(M) x 2(N); each wave: 32x64 = 2x4 accumulators of 16x16.
// Fragment lane map (16-bit A 16x32, ISA 7.12.2):
//   lane<16:  M=lane,    K = 0..7 (v0..3) and 16..23 (v4..7)
//   lane>=16: M=lane-16, K = 8..15 and 24..31
// ---------------------------------------------------------------------------
__global__ __launch_bounds__(256) void gemm_bf16_wmma(
    const __bf16* __restrict__ A, const __bf16* __restrict__ Bt,
    const float* __restrict__ bias, float* __restrict__ Cf,
    __bf16* __restrict__ Cb, int M, int N, int K, int relu) {
  __shared__ __align__(16) __bf16 As[2][128][40];  // 80B row stride (16B aligned)
  __shared__ __align__(16) __bf16 Bs[2][128][40];

  const int tid = threadIdx.x;
  const int w = tid >> 5, lane = tid & 31;
  const int hl = lane >> 4, ln = lane & 15;
  const int wm = w & 3, wn = w >> 2;
  const int m0 = blockIdx.y * 128, n0 = blockIdx.x * 128;

  v8f acc[2][4] = {};

  auto compute = [&](const __bf16(*Asl)[40], const __bf16(*Bsl)[40]) {
    v16bf af[2];
#pragma unroll
    for (int s = 0; s < 2; ++s) {
      const int r = wm * 32 + s * 16 + ln;
      af[s] = load_frag(&Asl[r][hl * 8], &Asl[r][16 + hl * 8]);
    }
#pragma unroll
    for (int j = 0; j < 4; ++j) {
      const int n = wn * 64 + j * 16 + ln;
      v16bf bf = load_frag(&Bsl[n][hl * 8], &Bsl[n][16 + hl * 8]);
#pragma unroll
      for (int s = 0; s < 2; ++s)
        acc[s][j] = __builtin_amdgcn_wmma_f32_16x16x32_bf16(
            false, af[s], false, bf, (short)0, acc[s][j], false, false);
    }
  };

  const int nIter = K >> 5;

#if USE_TDM
  // --- TDM double-buffered pipeline: DMA of tile i+1 overlaps WMMA of tile i.
  const bool issuer = (w == 0);
  if (issuer) {
    tdm_load_tile_bf16(A  + (size_t)m0 * K, (unsigned)(__UINTPTR_TYPE__)&As[0][0][0],
                       32u, 128u, (unsigned)K);
    tdm_load_tile_bf16(Bt + (size_t)n0 * K, (unsigned)(__UINTPTR_TYPE__)&Bs[0][0][0],
                       32u, 128u, (unsigned)K);
  }
  for (int i = 0; i < nIter; ++i) {
    const int cur = i & 1;
    if (issuer) __builtin_amdgcn_s_wait_tensorcnt(0);  // current tile landed
    __syncthreads();                                   // all waves see it; prev compute done
    if (issuer && (i + 1) < nIter) {
      const int k0n = (i + 1) << 5;
      tdm_load_tile_bf16(A  + (size_t)m0 * K + k0n,
                         (unsigned)(__UINTPTR_TYPE__)&As[cur ^ 1][0][0], 32u, 128u, (unsigned)K);
      tdm_load_tile_bf16(Bt + (size_t)n0 * K + k0n,
                         (unsigned)(__UINTPTR_TYPE__)&Bs[cur ^ 1][0][0], 32u, 128u, (unsigned)K);
    }
    compute(As[cur], Bs[cur]);
  }
#else
  // --- fallback: manual staging (host pass / toolchains without TDM builtin)
  for (int i = 0; i < nIter; ++i) {
    const int k0 = i << 5;
    __syncthreads();
#pragma unroll
    for (int it = 0; it < 2; ++it) {
      int c = tid + it * 256;
      int r = c >> 2, co = (c & 3) * 8;
      *(v8bf*)&As[0][r][co] = *(const v8bf*)(A  + (size_t)(m0 + r) * K + k0 + co);
      *(v8bf*)&Bs[0][r][co] = *(const v8bf*)(Bt + (size_t)(n0 + r) * K + k0 + co);
    }
    __syncthreads();
    if (k0 + 32 < K) {
      __builtin_prefetch(A  + (size_t)(m0 + (tid >> 1)) * K + (k0 + 32) + (tid & 1) * 16, 0, 1);
      __builtin_prefetch(Bt + (size_t)(n0 + (tid >> 1)) * K + (k0 + 32) + (tid & 1) * 16, 0, 1);
    }
    compute(As[0], Bs[0]);
  }
#endif

  // epilogue: C layout -> VGPR i holds rows i (lanes 0-15) / i+8 (lanes 16-31)
#pragma unroll
  for (int s = 0; s < 2; ++s) {
#pragma unroll
    for (int j = 0; j < 4; ++j) {
      const int gc = n0 + wn * 64 + j * 16 + ln;
      const float bval = bias ? bias[gc] : 0.f;
#pragma unroll
      for (int i = 0; i < 8; ++i) {
        const int gr = m0 + wm * 32 + s * 16 + i + 8 * hl;
        float v = acc[s][j][i] + bval;
        if (relu) v = fmaxf(v, 0.f);
        if (Cf) Cf[(size_t)gr * N + gc] = v;
        if (Cb) Cb[(size_t)gr * N + gc] = (__bf16)v;
      }
    }
  }
}

// ---------------------------------------------------------------------------
// Flash attention: grid (S/64, H, B), 128 threads = 4 waves.
// Each wave: 16 query rows x full Dh=64. Online softmax (scale 1/sqrt(D)=1/32).
// ---------------------------------------------------------------------------
__global__ __launch_bounds__(128) void flash_attn_bf16(
    const __bf16* __restrict__ qb, const __bf16* __restrict__ kb,
    const __bf16* __restrict__ vb, float* __restrict__ attn) {
  __shared__ __align__(16) __bf16 Qs[64][72];
  __shared__ __align__(16) __bf16 Ks[64][72];     // Ks[key][dh]  (B-frag for Q@K^T)
  __shared__ __align__(16) __bf16 Vt[64][72];     // Vt[dh][key]  (B-frag for P@V)
  __shared__ __align__(16) __bf16 Ps[4][16][72];  // per-wave P tile (A-frag relayout)

  const int b = blockIdx.z, h = blockIdx.y;
  const int q0 = blockIdx.x * 64;
  const int tid = threadIdx.x;
  const int w = tid >> 5, lane = tid & 31;
  const int hl = lane >> 4, ln = lane & 15;

  for (int c = tid; c < 512; c += 128) {  // Q block [64 x Dh]
    int r = c >> 3, co = (c & 7) * 8;
    const __bf16* src = qb + ((size_t)(b * SS + q0 + r)) * DD + h * DH + co;
    *(v8bf*)&Qs[r][co] = *(const v8bf*)src;
  }

  float mrow[8], lrow[8];
  v8f oacc[4] = {};
#pragma unroll
  for (int i = 0; i < 8; ++i) { mrow[i] = -1e30f; lrow[i] = 0.f; }

  for (int kt = 0; kt < SS / 64; ++kt) {
    __syncthreads();
    for (int c = tid; c < 512; c += 128) {  // stage K, V tiles (V transposed)
      int r = c >> 3, co = (c & 7) * 8;
      const size_t grow = (size_t)(b * SS + kt * 64 + r) * DD + h * DH + co;
      *(v8bf*)&Ks[r][co] = *(const v8bf*)(kb + grow);
      v8bf vv = *(const v8bf*)(vb + grow);
#pragma unroll
      for (int j = 0; j < 8; ++j) Vt[co + j][r] = vv[j];
    }
    __syncthreads();

    if (kt + 1 < SS / 64 && tid < 64) {  // global_prefetch_b8 next tiles
      const size_t nrow = (size_t)(b * SS + (kt + 1) * 64 + tid) * DD + h * DH;
      __builtin_prefetch(kb + nrow, 0, 1);
      __builtin_prefetch(vb + nrow, 0, 1);
    }

    // scores S = Q @ K^T  (4 key-subtiles x 2 K-steps over Dh)
    v8f sacc[4] = {};
#pragma unroll
    for (int kk = 0; kk < 2; ++kk) {
      const int qr = w * 16 + ln;
      v16bf af = load_frag(&Qs[qr][kk * 32 + hl * 8], &Qs[qr][kk * 32 + 16 + hl * 8]);
#pragma unroll
      for (int t = 0; t < 4; ++t) {
        const int kr = t * 16 + ln;
        v16bf bf = load_frag(&Ks[kr][kk * 32 + hl * 8], &Ks[kr][kk * 32 + 16 + hl * 8]);
        sacc[t] = __builtin_amdgcn_wmma_f32_16x16x32_bf16(
            false, af, false, bf, (short)0, sacc[t], false, false);
      }
    }

    // online softmax (rows live across 16-lane halves; shfl offsets 1..8)
    const float sc = 0.03125f;  // 1/sqrt(1024)
#pragma unroll
    for (int i = 0; i < 8; ++i) {
      float tmax = -1e30f;
#pragma unroll
      for (int t = 0; t < 4; ++t) tmax = fmaxf(tmax, sacc[t][i]);
      tmax *= sc;
      for (int o = 1; o < 16; o <<= 1) tmax = fmaxf(tmax, __shfl_xor(tmax, o, 32));
      const float mnew = fmaxf(mrow[i], tmax);
      const float fscale = __expf(mrow[i] - mnew);
      float psum = 0.f;
#pragma unroll
      for (int t = 0; t < 4; ++t) {
        float p = __expf(sacc[t][i] * sc - mnew);
        psum += p;
        Ps[w][i + 8 * hl][t * 16 + ln] = (__bf16)p;
      }
      for (int o = 1; o < 16; o <<= 1) psum += __shfl_xor(psum, o, 32);
      lrow[i] = lrow[i] * fscale + psum;
      mrow[i] = mnew;
#pragma unroll
      for (int d = 0; d < 4; ++d) oacc[d][i] = oacc[d][i] * fscale;
    }

    // O += P @ V  (per-wave Ps; DS ops same-wave are in-order)
#pragma unroll
    for (int kk = 0; kk < 2; ++kk) {
      v16bf af = load_frag(&Ps[w][ln][kk * 32 + hl * 8], &Ps[w][ln][kk * 32 + 16 + hl * 8]);
#pragma unroll
      for (int d = 0; d < 4; ++d) {
        const int n = d * 16 + ln;
        v16bf bf = load_frag(&Vt[n][kk * 32 + hl * 8], &Vt[n][kk * 32 + 16 + hl * 8]);
        oacc[d] = __builtin_amdgcn_wmma_f32_16x16x32_bf16(
            false, af, false, bf, (short)0, oacc[d], false, false);
      }
    }
  }

  // write O / l  as attn[b,h,s,dh] (f32)
#pragma unroll
  for (int d = 0; d < 4; ++d) {
#pragma unroll
    for (int i = 0; i < 8; ++i) {
      const int r = q0 + w * 16 + i + 8 * hl;
      const int dh = d * 16 + ln;
      attn[(((size_t)(b * HH + h)) * SS + r) * DH + dh] = oacc[d][i] / lrow[i];
    }
  }
}

// ---------------------------------------------------------------------------
// Reproduce reference transpose bug: [B,H,S,Dh] -> transpose(0,1,3,2) ->
// reshape(B,S,H*Dh). ctx[b,s,j] = attn[b, h, t, dh] with
//   f = s*D + j ; h = f/(Dh*S) ; dh = (f%(Dh*S))/S ; t = f%S
// ---------------------------------------------------------------------------
__global__ __launch_bounds__(256) void gather_ctx(const float* __restrict__ attn,
                                                  __bf16* __restrict__ ctx) {
  int idx = blockIdx.x * 256 + threadIdx.x;
  if (idx >= BB * SS * DD) return;
  const int b = idx / (SS * DD);
  const int f = idx - b * (SS * DD);  // == s*D + j
  const int h = f / (DH * SS);
  const int r2 = f - h * (DH * SS);
  const int dh = r2 / SS;
  const int t = r2 - dh * SS;
  ctx[idx] = (__bf16)attn[(((size_t)(b * HH + h)) * SS + t) * DH + dh];
}

// ---------------------------------------------------------------------------
// Fused LayerNorm + residual: out = LN(inp)*g + beta + res  (D = 1024)
// one block (256 thr, 8 waves) per row
// ---------------------------------------------------------------------------
__global__ __launch_bounds__(256) void ln_residual(
    const float* __restrict__ inp, const float* __restrict__ res,
    const float* __restrict__ g, const float* __restrict__ beta,
    float* __restrict__ outf, __bf16* __restrict__ outb) {
  const int row = blockIdx.x;
  const int tid = threadIdx.x;
  const float* xr = inp + (size_t)row * DD;
  float4 v = *(const float4*)(xr + tid * 4);

  __shared__ float red[8];
  float s = v.x + v.y + v.z + v.w;
  for (int o = 16; o > 0; o >>= 1) s += __shfl_xor(s, o, 32);
  if ((tid & 31) == 0) red[tid >> 5] = s;
  __syncthreads();
  float mu = 0.f;
#pragma unroll
  for (int i = 0; i < 8; ++i) mu += red[i];
  mu *= (1.f / DD);
  __syncthreads();

  float dx = v.x - mu, dy = v.y - mu, dz = v.z - mu, dw = v.w - mu;
  float s2 = dx * dx + dy * dy + dz * dz + dw * dw;
  for (int o = 16; o > 0; o >>= 1) s2 += __shfl_xor(s2, o, 32);
  if ((tid & 31) == 0) red[tid >> 5] = s2;
  __syncthreads();
  float var = 0.f;
#pragma unroll
  for (int i = 0; i < 8; ++i) var += red[i];
  var *= (1.f / DD);
  const float rstd = rsqrtf(var + 1e-5f);

  const float xs[4] = {v.x, v.y, v.z, v.w};
#pragma unroll
  for (int q = 0; q < 4; ++q) {
    const int c = tid * 4 + q;
    const float outv = (xs[q] - mu) * rstd * g[c] + beta[c] + res[(size_t)row * DD + c];
    if (outf) outf[(size_t)row * DD + c] = outv;
    if (outb) outb[(size_t)row * DD + c] = (__bf16)outv;
  }
}

// ---------------------------------------------------------------------------
// Host orchestration
// ---------------------------------------------------------------------------
extern "C" void kernel_launch(void* const* d_in, const int* in_sizes, int n_in,
                              void* d_out, int out_size, void* d_ws, size_t ws_size,
                              hipStream_t stream) {
  (void)in_sizes; (void)n_in; (void)out_size; (void)ws_size;
  const float* x  = (const float*)d_in[0];
  const float* wq = (const float*)d_in[1];
  const float* bq = (const float*)d_in[2];
  const float* wk = (const float*)d_in[3];
  const float* bk = (const float*)d_in[4];
  const float* wv = (const float*)d_in[5];
  const float* bv = (const float*)d_in[6];
  const float* wo = (const float*)d_in[7];
  const float* bo = (const float*)d_in[8];
  const float* ln1g = (const float*)d_in[9];
  const float* ln1b = (const float*)d_in[10];
  const float* ln2g = (const float*)d_in[11];
  const float* ln2b = (const float*)d_in[12];
  const float* w1 = (const float*)d_in[13];
  const float* w2 = (const float*)d_in[14];

  char* ws = (char*)d_ws;
  size_t off = 0;
  auto carve = [&](size_t bytes) -> void* {
    void* p = ws + off;
    off += (bytes + 255) & ~(size_t)255;
    return p;
  };
  __bf16* xb   = (__bf16*)carve((size_t)ROWS * DD * 2);
  __bf16* wqT  = (__bf16*)carve((size_t)DD * DD * 2);   // [N][K] transposed weights
  __bf16* wkT  = (__bf16*)carve((size_t)DD * DD * 2);
  __bf16* wvT  = (__bf16*)carve((size_t)DD * DD * 2);
  __bf16* woT  = (__bf16*)carve((size_t)DD * DD * 2);
  __bf16* w1T  = (__bf16*)carve((size_t)DD * FF * 2);
  __bf16* w2T  = (__bf16*)carve((size_t)FF * DD * 2);
  __bf16* qbuf = (__bf16*)carve((size_t)ROWS * DD * 2);
  __bf16* kbuf = (__bf16*)carve((size_t)ROWS * DD * 2);
  __bf16* vbuf = (__bf16*)carve((size_t)ROWS * DD * 2);
  float*  attn = (float*)carve((size_t)ROWS * DD * 4);  // reused for ffn output
  __bf16* ctxb = (__bf16*)carve((size_t)ROWS * DD * 2);
  float*  att  = (float*)carve((size_t)ROWS * DD * 4);
  float*  y1   = (float*)carve((size_t)ROWS * DD * 4);
  __bf16* y1b  = (__bf16*)carve((size_t)ROWS * DD * 2);
  __bf16* h1b  = (__bf16*)carve((size_t)ROWS * FF * 2);
  float*  ffn  = attn;  // lifetime-disjoint reuse

  // activations: plain convert; weights: transpose+convert (once, L2-resident)
  cvt_f32_bf16<<<(ROWS * DD / 4 + 255) / 256, 256, 0, stream>>>(x, xb, ROWS * DD);
  transpose_cvt_bf16<<<dim3(DD / 32, DD / 32), 256, 0, stream>>>(wq, wqT, DD, DD);
  transpose_cvt_bf16<<<dim3(DD / 32, DD / 32), 256, 0, stream>>>(wk, wkT, DD, DD);
  transpose_cvt_bf16<<<dim3(DD / 32, DD / 32), 256, 0, stream>>>(wv, wvT, DD, DD);
  transpose_cvt_bf16<<<dim3(DD / 32, DD / 32), 256, 0, stream>>>(wo, woT, DD, DD);
  transpose_cvt_bf16<<<dim3(DD / 32, FF / 32), 256, 0, stream>>>(w1, w1T, DD, FF);
  transpose_cvt_bf16<<<dim3(FF / 32, DD / 32), 256, 0, stream>>>(w2, w2T, FF, DD);

  // QKV projections: [4096,1024] @ [1024,1024] + bias -> bf16
  dim3 gQKV(DD / 128, ROWS / 128);
  gemm_bf16_wmma<<<gQKV, 256, 0, stream>>>(xb, wqT, bq, nullptr, qbuf, ROWS, DD, DD, 0);
  gemm_bf16_wmma<<<gQKV, 256, 0, stream>>>(xb, wkT, bk, nullptr, kbuf, ROWS, DD, DD, 0);
  gemm_bf16_wmma<<<gQKV, 256, 0, stream>>>(xb, wvT, bv, nullptr, vbuf, ROWS, DD, DD, 0);

  // flash attention per (b, h, 64-row query block)
  flash_attn_bf16<<<dim3(SS / 64, HH, BB), 128, 0, stream>>>(qbuf, kbuf, vbuf, attn);

  // reference's transpose bug + bf16 for the O-projection
  gather_ctx<<<(ROWS * DD + 255) / 256, 256, 0, stream>>>(attn, ctxb);

  // att = ctx @ wo + bo  (f32)
  gemm_bf16_wmma<<<gQKV, 256, 0, stream>>>(ctxb, woT, bo, att, nullptr, ROWS, DD, DD, 0);

  // y1 = LN(att)*g1 + b1 + x   (f32 + bf16 copy for FFN)
  ln_residual<<<ROWS, 256, 0, stream>>>(att, x, ln1g, ln1b, y1, y1b);

  // h1 = relu(y1 @ w1)  -> bf16 [4096, 4096]
  gemm_bf16_wmma<<<dim3(FF / 128, ROWS / 128), 256, 0, stream>>>(
      y1b, w1T, nullptr, nullptr, h1b, ROWS, FF, DD, 1);

  // ffn = h1 @ w2  (f32)
  gemm_bf16_wmma<<<gQKV, 256, 0, stream>>>(h1b, w2T, nullptr, ffn, nullptr, ROWS, DD, FF, 0);

  // out = LN(ffn)*g2 + b2 + y1
  ln_residual<<<ROWS, 256, 0, stream>>>(ffn, y1, ln2g, ln2b, (float*)d_out, nullptr);
}
